// PointGNNModel_36043365548043
// MI455X (gfx1250) — compile-verified, matrix operand order checked by reading, not run on hardware
//
#include <hip/hip_runtime.h>

// ---------------------------------------------------------------------------
// Types
// ---------------------------------------------------------------------------
typedef unsigned short bf16_t;
typedef unsigned int   u32;
typedef __bf16 v16bf __attribute__((ext_vector_type(16)));
typedef float  v8f   __attribute__((ext_vector_type(8)));
typedef unsigned u32x4 __attribute__((ext_vector_type(4)));

union FragU { v16bf v; u32x4 q[2]; u32 u[8]; };

static __device__ __forceinline__ bf16_t f2bf(float f) {
  u32 u = __float_as_uint(f);
  u32 r = u + 0x7FFFu + ((u >> 16) & 1u);   // round-to-nearest-even
  return (bf16_t)(r >> 16);
}
static __device__ __forceinline__ float bf2f(bf16_t b) {
  return __uint_as_float(((u32)b) << 16);
}

// CDNA5 WMMA: D = A(16x32 bf16) * B(32x16 bf16) + C(16x16 f32)
static __device__ __forceinline__ v8f wmma_bf16(v16bf a, v16bf b, v8f c) {
  return __builtin_amdgcn_wmma_f32_16x16x32_bf16(false, a, false, b,
                                                 (short)0, c, false, false);
}

// A-fragment loader: A is row-major bf16, 16 rows at Abase, stride strideH
// halfs.  Lane layout (ISA 7.12.2, 16-bit A 16x32):
//   row = lane%16, half = lane/16
//   VGPR j<4 : K = k0 + 8*half + 2j {,+1}
//   VGPR j>=4: K = k0 + 16 + 8*half + 2(j-4) {,+1}
// => two contiguous 16B chunks per lane.
static __device__ __forceinline__ v16bf load_a_frag(const bf16_t* Abase,
                                                    int strideH, int k0, int lane) {
  int row = lane & 15, half = lane >> 4;
  const bf16_t* p = Abase + (size_t)row * strideH + k0 + 8 * half;
  FragU f;
  f.q[0] = *(const u32x4*)p;           // halfs [0,8)
  f.q[1] = *(const u32x4*)(p + 16);    // halfs [16,24)
  return f.v;
}

// B-fragment loader: weights pre-swizzled into 1KB blocks per (ntile,kstep),
// each lane's 8 dwords contiguous at lane*32 bytes.
static __device__ __forceinline__ v16bf load_b_frag(const bf16_t* Wsw, int nt,
                                                    int ksteps, int ks, int lane) {
  const u32x4* q = (const u32x4*)Wsw + ((size_t)(nt * ksteps + ks)) * 64 + lane * 2;
  FragU f;
  f.q[0] = q[0];
  f.q[1] = q[1];
  return f.v;
}

// ---------------------------------------------------------------------------
// Weight swizzle: f32 [Kr,Nr] row-major -> bf16 fragment-order [Kp,Np]
// perm==1: K-order remap for f_w1 ([geo(3)|h(300)] -> [h(300)|geo(3)])
// ---------------------------------------------------------------------------
__global__ void swizzle_kernel(const float* __restrict__ W, int Kr, int Nr,
                               int Kp, int Np, bf16_t* __restrict__ out, int perm) {
  size_t tid = (size_t)blockIdx.x * 256 + threadIdx.x;  // one packed u32 each
  size_t total = (size_t)Kp * Np / 2;
  if (tid >= total) return;
  int blk = (int)(tid >> 8);      // 256 u32 per fragment block
  int within = (int)(tid & 255);
  int lane = within >> 3, j = within & 7;
  int ksteps = Kp >> 5;
  int ntile = blk / ksteps, kstep = blk % ksteps;
  int half = lane >> 4, n = ntile * 16 + (lane & 15);
  int kb = kstep * 32 + 8 * half + ((j < 4) ? 2 * j : 16 + 2 * (j - 4));
  u32 packed = 0;
  for (int s = 0; s < 2; ++s) {
    int k = kb + s;
    bf16_t v = 0;
    if (n < Nr) {
      if (perm == 1) {
        if (k < 300) k += 3;
        else if (k < 303) k -= 300;
        else k = -1;
      } else if (k >= Kr) k = -1;
      if (k >= 0) v = f2bf(W[(size_t)k * Nr + n]);
    }
    packed |= ((u32)v) << (16 * s);
  }
  ((u32*)out)[tid] = packed;
}

// ---------------------------------------------------------------------------
// Small utility kernels
// ---------------------------------------------------------------------------
__global__ void zero_f32_kernel(float* __restrict__ p, size_t n) {
  size_t i = (size_t)blockIdx.x * blockDim.x + threadIdx.x;
  if (i < n) p[i] = 0.0f;
}

__global__ void cvt_pad_kernel(const float* __restrict__ src, int sw, int s_real,
                               bf16_t* __restrict__ dst, int dw, int rows) {
  size_t tid = (size_t)blockIdx.x * blockDim.x + threadIdx.x;
  size_t total = (size_t)rows * dw;
  if (tid >= total) return;
  int c = (int)(tid % dw);
  size_t r = tid / dw;
  float v = (c < s_real) ? src[r * sw + c] : 0.0f;
  dst[tid] = f2bf(v);
}

// delta = relu(relu(h@w1+b1)@w2+b2) : [N,300]->[64]->[3]  (tiny, VALU)
__global__ __launch_bounds__(64) void delta_kernel(
    const float* __restrict__ h, const float* __restrict__ w1,
    const float* __restrict__ b1, const float* __restrict__ w2,
    const float* __restrict__ b2, float* __restrict__ delta) {
  __shared__ float hrow[300];
  __shared__ float t[64];
  int i = blockIdx.x, tid = threadIdx.x;
  for (int k = tid; k < 300; k += 64) hrow[k] = h[(size_t)i * 304 + k];
  __syncthreads();
  float acc = b1[tid];
  for (int k = 0; k < 300; ++k) acc += hrow[k] * w1[k * 64 + tid];
  t[tid] = fmaxf(acc, 0.0f);
  __syncthreads();
  if (tid < 4) {
    float a = 0.0f;
    if (tid < 3) {
      a = b2[tid];
      for (int k = 0; k < 64; ++k) a += t[k] * w2[k * 3 + tid];
      a = fmaxf(a, 0.0f);
    }
    delta[(size_t)i * 4 + tid] = a;
  }
}

// final head: y = t3[:, :300] @ out_w + out_b  (tiny, VALU)
__global__ __launch_bounds__(32) void head_out_kernel(
    const bf16_t* __restrict__ t3, const float* __restrict__ w,
    const float* __restrict__ b, float* __restrict__ y) {
  int lane = threadIdx.x;
  int row = blockIdx.x * 16 + (lane >> 1);
  int c = lane & 1;
  float acc = b[c];
  const bf16_t* tr = t3 + (size_t)row * 320;
  for (int k = 0; k < 300; ++k) acc += bf2f(tr[k]) * w[k * 2 + c];
  y[(size_t)row * 2 + c] = acc;
}

// ---------------------------------------------------------------------------
// Generic node GEMM: out = relu(A_bf16[M,Kp] @ Wsw + bias)
//   optional: residual+2nd relu (PointGNN update), f32 mirror store.
// One wave per 16x16 output tile; grid = (M/16, Np/16).
// ---------------------------------------------------------------------------
template <bool RESID, bool F32OUT>
__global__ __launch_bounds__(32) void gemm_kernel(
    const bf16_t* __restrict__ A, int Kp, const bf16_t* __restrict__ Wsw,
    const float* __restrict__ bias, int n_real, bf16_t* __restrict__ outBF,
    int Np, float* __restrict__ outF, int f32W, const float* __restrict__ res,
    int resW) {
  int lane = threadIdx.x;
  int rt = blockIdx.x, nt = blockIdx.y;
  int row0 = rt * 16;
  int ksteps = Kp >> 5;
  v8f acc = {0.f, 0.f, 0.f, 0.f, 0.f, 0.f, 0.f, 0.f};
  const bf16_t* Abase = A + (size_t)row0 * Kp;
  for (int ks = 0; ks < ksteps; ++ks) {
    v16bf a = load_a_frag(Abase, Kp, ks * 32, lane);
    v16bf b = load_b_frag(Wsw, nt, ksteps, ks, lane);
    acc = wmma_bf16(a, b, acc);
  }
  int col = nt * 16 + (lane & 15);
  int half = lane >> 4;
  float bv = (col < n_real) ? bias[col] : 0.0f;
#pragma unroll
  for (int r = 0; r < 8; ++r) {
    int row = row0 + r + 8 * half;
    float v = fmaxf(acc[r] + bv, 0.0f);
    if (RESID) {
      if (col < n_real) v = fmaxf(res[(size_t)row * resW + col] + v, 0.0f);
    }
    outBF[(size_t)row * Np + col] = f2bf(v);
    if (F32OUT) {
      if (col < n_real) outF[(size_t)row * f32W + col] = v;
    }
  }
}

// ---------------------------------------------------------------------------
// Fused edge kernel: per 16-edge tile (one wave):
//   e[16,320] = [h_bf[src] (300) | pos_src-pos_dst+delta_dst (3) | 0...]
//   t1 = relu(e @ f_w1 + b1)         (LDS)
//   m  = relu(t1 @ f_w2 + b2)        (registers)
//   agg[dst] = max(agg[dst], m)      (atomic max on bits, m >= 0)
// ---------------------------------------------------------------------------
__global__ __launch_bounds__(32) void edge_kernel(
    const int* __restrict__ ei, const float* __restrict__ pos,
    const float* __restrict__ delta, const bf16_t* __restrict__ hbf,
    const bf16_t* __restrict__ w1sw, const float* __restrict__ b1,
    const bf16_t* __restrict__ w2sw, const float* __restrict__ b2,
    float* __restrict__ agg) {
  __shared__ __align__(16) bf16_t eT[16 * 320];
  __shared__ __align__(16) bf16_t t1T[16 * 320];
  __shared__ int srcL[16], dstL[16];
  const int lane = threadIdx.x;
  const int e0 = blockIdx.x * 16;
  if (lane < 16) {
    srcL[lane] = ei[e0 + lane];
    dstL[lane] = ei[160000 + e0 + lane];
  }
  __syncthreads();
  {  // copy h row (320 halfs incl. zero pads) -> e cols [0,320)
    int row = lane >> 1, sub = lane & 1;
    const u32x4* sp = (const u32x4*)(hbf + (size_t)srcL[row] * 320) + sub * 20;
    u32x4* dp = (u32x4*)(eT + row * 320) + sub * 20;
#pragma unroll
    for (int w = 0; w < 20; ++w) dp[w] = sp[w];
  }
  __syncthreads();
  if (lane < 16) {  // geometry into cols 300..302 (cols 303..319 stay 0)
    int s = srcL[lane], d = dstL[lane];
#pragma unroll
    for (int c = 0; c < 3; ++c) {
      float g = pos[s * 3 + c] - pos[d * 3 + c] + delta[(size_t)d * 4 + c];
      eT[lane * 320 + 300 + c] = f2bf(g);
    }
  }
  __syncthreads();

  const int KS = 10;  // 320/32
#pragma unroll 1
  for (int nt = 0; nt < 20; ++nt) {
    v8f acc = {0.f, 0.f, 0.f, 0.f, 0.f, 0.f, 0.f, 0.f};
#pragma unroll
    for (int ks = 0; ks < KS; ++ks) {
      v16bf a = load_a_frag(eT, 320, ks * 32, lane);
      v16bf b = load_b_frag(w1sw, nt, KS, ks, lane);
      acc = wmma_bf16(a, b, acc);
    }
    int col = nt * 16 + (lane & 15), half = lane >> 4;
    float bv = (col < 300) ? b1[col] : 0.0f;
#pragma unroll
    for (int r = 0; r < 8; ++r) {
      float v = fmaxf(acc[r] + bv, 0.0f);
      t1T[(r + 8 * half) * 320 + col] = f2bf(v);
    }
  }
  __syncthreads();

#pragma unroll 1
  for (int nt = 0; nt < 20; ++nt) {
    v8f acc = {0.f, 0.f, 0.f, 0.f, 0.f, 0.f, 0.f, 0.f};
#pragma unroll
    for (int ks = 0; ks < KS; ++ks) {
      v16bf a = load_a_frag(t1T, 320, ks * 32, lane);
      v16bf b = load_b_frag(w2sw, nt, KS, ks, lane);
      acc = wmma_bf16(a, b, acc);
    }
    int col = nt * 16 + (lane & 15), half = lane >> 4;
    if (col < 300) {
      float bv = b2[col];
#pragma unroll
      for (int r = 0; r < 8; ++r) {
        float v = fmaxf(acc[r] + bv, 0.0f);
        int row = r + 8 * half;
        atomicMax((int*)&agg[(size_t)dstL[row] * 304 + col], __float_as_int(v));
      }
    }
  }
}

// ---------------------------------------------------------------------------
// Host orchestration
// ---------------------------------------------------------------------------
extern "C" void kernel_launch(void* const* d_in, const int* in_sizes, int n_in,
                              void* d_out, int out_size, void* d_ws, size_t ws_size,
                              hipStream_t stream) {
  (void)in_sizes; (void)n_in; (void)out_size; (void)ws_size;
  const int N = 10000, E = 160000, L = 2;
  const float* x      = (const float*)d_in[0];
  const float* pos    = (const float*)d_in[1];
  const int*   ei     = (const int*)d_in[2];
  const float* emb_w1 = (const float*)d_in[3];
  const float* emb_b1 = (const float*)d_in[4];
  const float* emb_w2 = (const float*)d_in[5];
  const float* emb_b2 = (const float*)d_in[6];
  const float* h_w1   = (const float*)d_in[7];
  const float* h_b1   = (const float*)d_in[8];
  const float* h_w2   = (const float*)d_in[9];
  const float* h_b2   = (const float*)d_in[10];
  const float* f_w1   = (const float*)d_in[11];
  const float* f_b1   = (const float*)d_in[12];
  const float* f_w2   = (const float*)d_in[13];
  const float* f_b2   = (const float*)d_in[14];
  const float* g_w1   = (const float*)d_in[15];
  const float* g_b1   = (const float*)d_in[16];
  const float* g_w2   = (const float*)d_in[17];
  const float* g_b2   = (const float*)d_in[18];
  const float* fc1_w  = (const float*)d_in[19];
  const float* fc1_b  = (const float*)d_in[20];
  const float* fc2_w  = (const float*)d_in[21];
  const float* fc2_b  = (const float*)d_in[22];
  const float* fc3_w  = (const float*)d_in[23];
  const float* fc3_b  = (const float*)d_in[24];
  const float* out_w  = (const float*)d_in[25];
  const float* out_b  = (const float*)d_in[26];
  float* y = (float*)d_out;

  // workspace layout
  char* ws = (char*)d_ws;
  size_t off = 0;
  auto alloc = [&](size_t b) { size_t r = off; off = (off + b + 255) & ~(size_t)255; return r; };
  bf16_t* emb1sw = (bf16_t*)(ws + alloc((size_t)32 * 128 * 2));
  bf16_t* emb2sw = (bf16_t*)(ws + alloc((size_t)128 * 320 * 2));
  bf16_t* fw1sw[2], *fw2sw[2], *gw1sw[2], *gw2sw[2];
  for (int l = 0; l < L; ++l) fw1sw[l] = (bf16_t*)(ws + alloc((size_t)320 * 320 * 2));
  for (int l = 0; l < L; ++l) fw2sw[l] = (bf16_t*)(ws + alloc((size_t)320 * 320 * 2));
  for (int l = 0; l < L; ++l) gw1sw[l] = (bf16_t*)(ws + alloc((size_t)320 * 320 * 2));
  for (int l = 0; l < L; ++l) gw2sw[l] = (bf16_t*)(ws + alloc((size_t)320 * 320 * 2));
  bf16_t* fc1sw = (bf16_t*)(ws + alloc((size_t)320 * 1024 * 2));
  bf16_t* fc2sw = (bf16_t*)(ws + alloc((size_t)1024 * 512 * 2));
  bf16_t* fc3sw = (bf16_t*)(ws + alloc((size_t)512 * 320 * 2));
  bf16_t* h_bf  = (bf16_t*)(ws + alloc((size_t)N * 320 * 2));
  float*  h_f32 = (float*)(ws + alloc((size_t)N * 304 * 4));
  float*  dlt   = (float*)(ws + alloc((size_t)N * 4 * 4));
  float*  agg   = (float*)(ws + alloc((size_t)N * 304 * 4));
  // aliased regions (lifetimes disjoint): S1 = te | t_g1 | fc1-out | fc3-out
  bf16_t* S1 = (bf16_t*)(ws + alloc((size_t)N * 1024 * 2));
  // S2 = xb | agg_bf | fc2-out
  bf16_t* S2 = (bf16_t*)(ws + alloc((size_t)N * 512 * 2));

  auto swz = [&](const float* W, int Kr, int Nr, int Kp, int Np, bf16_t* dst, int perm) {
    size_t tot = (size_t)Kp * Np / 2;
    swizzle_kernel<<<(unsigned)((tot + 255) / 256), 256, 0, stream>>>(W, Kr, Nr, Kp, Np, dst, perm);
  };

  // --- weight prep (bf16 + fragment swizzle) ---
  swz(emb_w1, 16, 128, 32, 128, emb1sw, 0);
  swz(emb_w2, 128, 300, 128, 320, emb2sw, 0);
  for (int l = 0; l < L; ++l) {
    swz(f_w1 + (size_t)l * 303 * 300, 303, 300, 320, 320, fw1sw[l], 1);
    swz(f_w2 + (size_t)l * 300 * 300, 300, 300, 320, 320, fw2sw[l], 0);
    swz(g_w1 + (size_t)l * 300 * 300, 300, 300, 320, 320, gw1sw[l], 0);
    swz(g_w2 + (size_t)l * 300 * 300, 300, 300, 320, 320, gw2sw[l], 0);
  }
  swz(fc1_w, 300, 1024, 320, 1024, fc1sw, 0);
  swz(fc2_w, 1024, 512, 1024, 512, fc2sw, 0);
  swz(fc3_w, 512, 300, 512, 320, fc3sw, 0);

  // --- embedding: h = relu(relu(x@W1+b1)@W2+b2) ---
  {
    size_t tot = (size_t)N * 32;
    cvt_pad_kernel<<<(unsigned)((tot + 255) / 256), 256, 0, stream>>>(x, 16, 16, S2, 32, N);
  }
  gemm_kernel<false, false><<<dim3(625, 8), 32, 0, stream>>>(
      S2, 32, emb1sw, emb_b1, 128, S1, 128, nullptr, 0, nullptr, 0);
  gemm_kernel<false, true><<<dim3(625, 20), 32, 0, stream>>>(
      S1, 128, emb2sw, emb_b2, 300, h_bf, 320, h_f32, 304, nullptr, 0);

  // --- GNN layers ---
  for (int l = 0; l < L; ++l) {
    delta_kernel<<<N, 64, 0, stream>>>(h_f32, h_w1 + (size_t)l * 300 * 64,
                                       h_b1 + (size_t)l * 64, h_w2 + (size_t)l * 64 * 3,
                                       h_b2 + (size_t)l * 3, dlt);
    {
      size_t tot = (size_t)N * 304;
      zero_f32_kernel<<<(unsigned)((tot + 255) / 256), 256, 0, stream>>>(agg, tot);
    }
    edge_kernel<<<E / 16, 32, 0, stream>>>(ei, pos, dlt, h_bf, fw1sw[l],
                                           f_b1 + (size_t)l * 300, fw2sw[l],
                                           f_b2 + (size_t)l * 300, agg);
    {
      size_t tot = (size_t)N * 320;
      cvt_pad_kernel<<<(unsigned)((tot + 255) / 256), 256, 0, stream>>>(agg, 304, 304, S2, 320, N);
    }
    gemm_kernel<false, false><<<dim3(625, 20), 32, 0, stream>>>(
        S2, 320, gw1sw[l], g_b1 + (size_t)l * 300, 300, S1, 320, nullptr, 0, nullptr, 0);
    gemm_kernel<true, true><<<dim3(625, 20), 32, 0, stream>>>(
        S1, 320, gw2sw[l], g_b2 + (size_t)l * 300, 300, h_bf, 320, h_f32, 304, h_f32, 304);
  }

  // --- dense head ---
  gemm_kernel<false, false><<<dim3(625, 64), 32, 0, stream>>>(
      h_bf, 320, fc1sw, fc1_b, 1024, S1, 1024, nullptr, 0, nullptr, 0);
  gemm_kernel<false, false><<<dim3(625, 32), 32, 0, stream>>>(
      S1, 1024, fc2sw, fc2_b, 512, S2, 512, nullptr, 0, nullptr, 0);
  gemm_kernel<false, false><<<dim3(625, 20), 32, 0, stream>>>(
      S2, 512, fc3sw, fc3_b, 300, S1, 320, nullptr, 0, nullptr, 0);
  head_out_kernel<<<625, 32, 0, stream>>>(S1, out_w, out_b, y);
}
